// DualCollborativeMultiLevelEncoder_28372553957975
// MI455X (gfx1250) — matrix operand order repaired and checked by reading, not compiled, not run on hardware
//
#include <hip/hip_runtime.h>
#include <hip/hip_bf16.h>
#include <math.h>

// ---------------- CDNA5 types ----------------
typedef __bf16 bf16_t;
typedef __attribute__((ext_vector_type(16))) __bf16 v16bf;
typedef __attribute__((ext_vector_type(8)))  __bf16 v8bf;
typedef __attribute__((ext_vector_type(8)))  float  v8f;
typedef __attribute__((ext_vector_type(4)))  unsigned int v4u;
typedef __attribute__((ext_vector_type(4)))  int v4i;
typedef __attribute__((ext_vector_type(8)))  int v8i;

#define D_MODEL 512
#define H 8
#define DKH 64
#define DFF 2048
#define NL 3
#define BS 32
#define NR 100
#define NG 144
#define NALL 244
#define DIM_G 64

__device__ __forceinline__ bf16_t f2b(float x) { return (bf16_t)x; }

#if defined(__has_builtin)
#if __has_builtin(__builtin_amdgcn_tensor_load_to_lds)
#define HAVE_TDM 1
#endif
#endif

// =====================================================================
// Batched GEMM: C[M,N] = alpha * A[M,K] @ op(B) (+bias)(+relu)
// Block = 128 threads = 4 wave32s. Block tile 64x64:
//   wave w owns N columns [w*16, w*16+16); 4 M sub-tiles => 4 accumulators
//   => 4 v_wmma_f32_16x16x32_bf16 per 32-deep K step, B fragment reused.
// LDS: A row-major [64][32] (pad 40), B transposed [N=64][K=32] (pad 40)
// so fragments are contiguous ds_load_b128 pairs.  Global->LDS staging is
// done in 16-byte chunks (global_load_b128 -> ds_store_b128) whenever the
// leading dimension permits.
// =====================================================================
#define SAS 40
#define SBS 40

__global__ void k_gemm_wmma(
    const bf16_t* __restrict__ A, long strideA, int lda,
    const bf16_t* __restrict__ B, long strideB, int ldb, int transB,
    float* __restrict__ C32, bf16_t* __restrict__ Cbf, long strideC, int ldc,
    const float* __restrict__ bias, float alpha, int relu,
    int M, int N, int K)
{
  __align__(16) __shared__ bf16_t sA[64 * SAS];
  __align__(16) __shared__ bf16_t sBt[64 * SBS];

  const int bz   = blockIdx.z;
  const bf16_t* Ab = A + (long)bz * strideA;
  const bf16_t* Bb = B + (long)bz * strideB;
  const int row0 = blockIdx.y * 64;
  const int colB = blockIdx.x * 64;
  const int tid  = threadIdx.x;
  const int wave = tid >> 5;
  const int lane = tid & 31;
  const int half = lane >> 4;
  const int l16  = lane & 15;
  const bool vecA = ((lda & 7) == 0);
  const bool vecB = ((ldb & 7) == 0);

  v8f acc[4] = {{}, {}, {}, {}};
  union V16 { v16bf v; v8bf h[2]; };

  for (int k0 = 0; k0 < K; k0 += 32) {
    // ---- stage A 64x32: 256 chunks of 8 bf16 (16B), 2 per thread ----
    #pragma unroll
    for (int ch = tid; ch < 256; ch += 128) {
      int r = ch >> 2;             // 0..63
      int c = (ch & 3) << 3;       // 0,8,16,24
      int gr = row0 + r; if (gr >= M) gr = M - 1;
      int gk = k0 + c;
      v8bf vv;
      if (vecA && gk + 8 <= K) {
        vv = *(const v8bf*)&Ab[(long)gr * lda + gk];       // global_load_b128
      } else {
        #pragma unroll
        for (int e = 0; e < 8; ++e)
          vv[e] = (gk + e < K) ? Ab[(long)gr * lda + gk + e] : f2b(0.f);
      }
      *(v8bf*)&sA[r * SAS + c] = vv;                        // ds_store_b128
    }
    // ---- stage B transposed to [n][k] ----
    if (transB) {
      #pragma unroll
      for (int ch = tid; ch < 256; ch += 128) {
        int nn = ch >> 2;          // 0..63
        int c  = (ch & 3) << 3;    // 0,8,16,24
        int gn = colB + nn; if (gn >= N) gn = N - 1;
        int gk = k0 + c;
        v8bf vv;
        if (vecB && gk + 8 <= K) {
          vv = *(const v8bf*)&Bb[(long)gn * ldb + gk];
        } else {
          #pragma unroll
          for (int e = 0; e < 8; ++e)
            vv[e] = (gk + e < K) ? Bb[(long)gn * ldb + gk + e] : f2b(0.f);
        }
        *(v8bf*)&sBt[nn * SBS + c] = vv;
      }
    } else {
      #pragma unroll
      for (int ch = tid; ch < 256; ch += 128) {
        int kk  = ch >> 3;         // 0..31
        int n8  = (ch & 7) << 3;   // 0,8,...,56
        int gk  = k0 + kk;
        int gn0 = colB + n8;
        v8bf vv;
        if (vecB && gk < K && gn0 + 8 <= N) {
          vv = *(const v8bf*)&Bb[(long)gk * ldb + gn0];     // coalesced along N
        } else {
          #pragma unroll
          for (int e = 0; e < 8; ++e) {
            int gn = gn0 + e; if (gn >= N) gn = N - 1;
            vv[e] = (gk < K) ? Bb[(long)gk * ldb + gn] : f2b(0.f);
          }
        }
        #pragma unroll
        for (int e = 0; e < 8; ++e) sBt[(n8 + e) * SBS + kk] = vv[e];  // transpose scatter
      }
    }
    if (k0 + 32 < K)
      __builtin_prefetch(&Ab[(long)row0 * lda + k0 + 32], 0, 1);  // global_prefetch_b8
    __syncthreads();

    // B fragment (lanes 0-15: K{0..7,16..23}; lanes 16-31: K{8..15,24..31}),
    // loaded once per wave per k-step, reused by all 4 M sub-tiles.
    V16 bf_;
    {
      const v8bf* pb = (const v8bf*)&sBt[(wave * 16 + l16) * SBS + (half ? 8 : 0)];
      bf_.h[0] = pb[0];
      bf_.h[1] = pb[2];
    }
    #pragma unroll
    for (int m = 0; m < 4; ++m) {
      // A fragment (lanes 0-15: K{0..7,16..23}; lanes 16-31: K{8..15,24..31})
      V16 af;
      const v8bf* pa = (const v8bf*)&sA[(m * 16 + l16) * SAS + (half ? 8 : 0)];
      af.h[0] = pa[0];
      af.h[1] = pa[2];   // +16 elements
      acc[m] = __builtin_amdgcn_wmma_f32_16x16x32_bf16(
          false, af.v, false, bf_.v, (short)0, acc[m], false, false);
    }
    __syncthreads();
  }

  // C layout: VGPR v -> M = v + 8*half, N = l16 (within wave's 16-col strip)
  float*  Cb = C32 ? C32 + (long)bz * strideC : nullptr;
  bf16_t* Cf = Cbf ? Cbf + (long)bz * strideC : nullptr;
  const int c = colB + wave * 16 + l16;
  #pragma unroll
  for (int m = 0; m < 4; ++m) {
    #pragma unroll
    for (int v = 0; v < 8; ++v) {
      int r = row0 + m * 16 + v + (half ? 8 : 0);
      if (r < M && c < N) {
        float val = acc[m][v] * alpha;
        if (bias) val += bias[c];
        if (relu) val = fmaxf(val, 0.f);
        long idx = (long)r * ldc + c;
        if (Cb) Cb[idx] = val;
        if (Cf) Cf[idx] = f2b(val);
      }
    }
  }
}

// ---------------- elementwise / packing kernels ----------------
__global__ void k_f2bf(const float* __restrict__ x, bf16_t* __restrict__ y, long n) {
  long i = (long)blockIdx.x * blockDim.x + threadIdx.x;
  if (i < n) y[i] = f2b(x[i]);
}
__global__ void k_add_f2bf(const float* __restrict__ a, const float* __restrict__ p,
                           bf16_t* __restrict__ y, long n) {
  long i = (long)blockIdx.x * blockDim.x + threadIdx.x;
  if (i < n) y[i] = f2b(a[i] + p[i]);
}
// X:[BS,n,512] fp32 -> Y:[BS*H,n,64] bf16
__global__ void k_split_heads(const float* __restrict__ X, bf16_t* __restrict__ Y, int n) {
  long i = (long)blockIdx.x * blockDim.x + threadIdx.x;
  long total = (long)BS * H * n * DKH;
  if (i >= total) return;
  int d = (int)(i & 63); long t = i >> 6;
  int r = (int)(t % n); t /= n;
  int h = (int)(t % H); int b = (int)(t / H);
  Y[i] = f2b(X[(((long)b * n + r) << 9) + h * DKH + d]);
}
// X:[BS*H,n,64] fp32 -> Y:[BS,n,512] bf16
__global__ void k_merge_heads(const float* __restrict__ X, bf16_t* __restrict__ Y, int n) {
  long i = (long)blockIdx.x * blockDim.x + threadIdx.x;
  long total = (long)BS * n * D_MODEL;
  if (i >= total) return;
  int c = (int)(i & 511); long t = i >> 9;
  int r = (int)(t % n); int b = (int)(t / n);
  int h = c >> 6; int d = c & 63;
  Y[i] = f2b(X[(((long)(b * H + h) * n + r) << 6) + d]);
}
// concat along node axis: a[BS,na,512] ++ b[BS,nb,512] -> y[BS,na+nb,512]
__global__ void k_concat2(const float* __restrict__ a, int na,
                          const float* __restrict__ b, int nb,
                          float* __restrict__ y) {
  long i = (long)blockIdx.x * blockDim.x + threadIdx.x;
  int nt = na + nb;
  long total = (long)BS * nt * D_MODEL;
  if (i >= total) return;
  int c = (int)(i & 511); long t = i >> 9;
  int r = (int)(t % nt); int bz = (int)(t / nt);
  y[i] = (r < na) ? a[((long)bz * na + r) * D_MODEL + c]
                  : b[((long)bz * nb + (r - na)) * D_MODEL + c];
}

// ---------------- relational embedding -> log(relu(gw)) bias ----------------
__global__ void k_gw_bias(const float* __restrict__ boxes,
                          const float* __restrict__ WGw,
                          const float* __restrict__ WGb,
                          float* __restrict__ out) {
  long idx = (long)blockIdx.x * blockDim.x + threadIdx.x;
  long total = (long)BS * NALL * NALL;
  if (idx >= total) return;
  int j = (int)(idx % NALL); long t = idx / NALL;
  int i = (int)(t % NALL);   int b = (int)(t / NALL);
  const float* bi = boxes + ((long)b * NALL + i) * 4;
  const float* bj = boxes + ((long)b * NALL + j) * 4;
  float cxi = (bi[0] + bi[2]) * 0.5f, cyi = (bi[1] + bi[3]) * 0.5f;
  float wi  = bi[2] - bi[0] + 1.f,    hi  = bi[3] - bi[1] + 1.f;
  float cxj = (bj[0] + bj[2]) * 0.5f, cyj = (bj[1] + bj[3]) * 0.5f;
  float wj  = bj[2] - bj[0] + 1.f,    hj  = bj[3] - bj[1] + 1.f;
  float pos[4];
  pos[0] = logf(fmaxf(fabsf((cxi - cxj) / wi), 0.001f));
  pos[1] = logf(fmaxf(fabsf((cyi - cyj) / hi), 0.001f));
  pos[2] = logf(wi / wj);
  pos[3] = logf(hi / hj);
  float acch[H];
  #pragma unroll
  for (int h = 0; h < H; ++h) acch[h] = WGb[h];
  for (int t32 = 0; t32 < 32; ++t32) {
    int p = t32 >> 3, f = t32 & 7;
    float dim = __powf(1000.f, (float)f * 0.125f);
    float m = pos[p] * 100.f / dim;
    float s, c; __sincosf(m, &s, &c);
    #pragma unroll
    for (int h = 0; h < H; ++h)
      acch[h] += s * WGw[h * DIM_G + t32] + c * WGw[h * DIM_G + 32 + t32];
  }
  for (int h = 0; h < H; ++h) {
    float g = fmaxf(acch[h], 0.f);
    out[((((long)b * H + h) * NALL) + i) * NALL + j] = logf(fmaxf(g, 1e-6f));
  }
}

// ---------------- masks ----------------
__global__ void k_feat_mask(const float* __restrict__ x, unsigned char* __restrict__ m) {
  int row = blockIdx.x;
  __shared__ int cnt;
  if (threadIdx.x == 0) cnt = 0;
  __syncthreads();
  int any = 0;
  for (int d = threadIdx.x; d < D_MODEL; d += blockDim.x)
    if (x[(long)row * D_MODEL + d] == 0.f) any = 1;
  if (any) atomicOr(&cnt, 1);
  __syncthreads();
  if (threadIdx.x == 0) m[row] = (unsigned char)(cnt ? 1 : 0);
}
__global__ void k_mask_ra(const int* __restrict__ aligns, unsigned char* __restrict__ ra) {
  long i = (long)blockIdx.x * blockDim.x + threadIdx.x;
  long total = (long)BS * NR * NALL;
  if (i >= total) return;
  int j = (int)(i % NALL); long t = i / NALL;
  int q = (int)(t % NR); int b = (int)(t / NR);
  unsigned char m;
  if (j < NR) m = (q != j);
  else        m = (aligns[((long)b * NR + q) * NG + (j - NR)] == 0);
  ra[i] = m;
}
__global__ void k_mask_ga(const int* __restrict__ aligns, unsigned char* __restrict__ ga) {
  long i = (long)blockIdx.x * blockDim.x + threadIdx.x;
  long total = (long)BS * NG * NALL;
  if (i >= total) return;
  int j = (int)(i % NALL); long t = i / NALL;
  int g = (int)(t % NG); int b = (int)(t / NG);
  unsigned char m;
  if (j < NR) m = (aligns[((long)b * NR + j) * NG + g] == 0);
  else        m = ((j - NR) != g);
  ga[i] = m;
}

// ---------------- softmax (adds gw log-bias + mask) ----------------
__global__ void k_softmax_att(const float* __restrict__ scores, long sstride,
                              const float* __restrict__ gwb, int qoff, int koff,
                              const unsigned char* __restrict__ mask, int mrow, long mbatch,
                              bf16_t* __restrict__ att, int nq, int nk) {
  int bh = blockIdx.y, q = blockIdx.x;
  int b = bh / H, h = bh % H;
  const float* srow = scores + (long)bh * sstride + (long)q * nk;
  const float* grow = gwb + ((((long)b * H + h) * NALL) + qoff + q) * NALL + koff;
  const unsigned char* mp = mask + (long)b * mbatch + (long)q * mrow;
  __shared__ float red[256];
  int t = threadIdx.x;
  float mx = -1e30f;
  for (int j = t; j < nk; j += 256) {
    float s = srow[j] + grow[j];
    if (mp[j]) s = -1e9f;
    mx = fmaxf(mx, s);
  }
  red[t] = mx; __syncthreads();
  for (int o = 128; o > 0; o >>= 1) { if (t < o) red[t] = fmaxf(red[t], red[t + o]); __syncthreads(); }
  mx = red[0]; __syncthreads();
  float sum = 0.f;
  for (int j = t; j < nk; j += 256) {
    float s = srow[j] + grow[j];
    if (mp[j]) s = -1e9f;
    sum += __expf(s - mx);
  }
  red[t] = sum; __syncthreads();
  for (int o = 128; o > 0; o >>= 1) { if (t < o) red[t] += red[t + o]; __syncthreads(); }
  float inv = 1.f / red[0];
  bf16_t* arow = att + (long)bh * sstride + (long)q * nk;
  for (int j = t; j < nk; j += 256) {
    float s = srow[j] + grow[j];
    if (mp[j]) s = -1e9f;
    arow[j] = f2b(__expf(s - mx) * inv);
  }
}

// ---------------- LayerNorm(x + r1 + r2) ----------------
__global__ void k_ln(const float* __restrict__ x, const float* __restrict__ r1,
                     const float* __restrict__ r2,
                     const float* __restrict__ g, const float* __restrict__ bb,
                     float* __restrict__ y32, bf16_t* __restrict__ ybf) {
  long row = blockIdx.x;
  int t = threadIdx.x;
  long o0 = row * D_MODEL + t, o1 = o0 + 256;
  float e0 = x[o0], e1 = x[o1];
  if (r1) { e0 += r1[o0]; e1 += r1[o1]; }
  if (r2) { e0 += r2[o0]; e1 += r2[o1]; }
  __shared__ float red[256];
  red[t] = e0 + e1; __syncthreads();
  for (int s = 128; s > 0; s >>= 1) { if (t < s) red[t] += red[t + s]; __syncthreads(); }
  float mean = red[0] * (1.f / 512.f); __syncthreads();
  float d0 = e0 - mean, d1 = e1 - mean;
  red[t] = d0 * d0 + d1 * d1; __syncthreads();
  for (int s = 128; s > 0; s >>= 1) { if (t < s) red[t] += red[t + s]; __syncthreads(); }
  float inv = rsqrtf(red[0] * (1.f / 512.f) + 1e-5f);
  float v0 = d0 * inv * g[t] + bb[t];
  float v1 = d1 * inv * g[t + 256] + bb[t + 256];
  if (y32) { y32[o0] = v0; y32[o1] = v1; }
  if (ybf) { ybf[o0] = f2b(v0); ybf[o1] = f2b(v1); }
}

// ---------------- output assembly ----------------
__global__ void k_write_out(const float* __restrict__ reg, const float* __restrict__ grd,
                            float* __restrict__ outs, int layer) {
  long i = (long)blockIdx.x * blockDim.x + threadIdx.x;
  long total = (long)BS * NALL * D_MODEL;
  if (i >= total) return;
  int c = (int)(i & 511); long t = i >> 9;
  int r = (int)(t % NALL); int b = (int)(t / NALL);
  float v = (r < NR) ? reg[((long)b * NR + r) * D_MODEL + c]
                     : grd[((long)b * NG + (r - NR)) * D_MODEL + c];
  outs[(((long)b * NL + layer) * NALL + r) * D_MODEL + c] = v;
}
__global__ void k_write_mask(const unsigned char* __restrict__ mr,
                             const unsigned char* __restrict__ mg,
                             float* __restrict__ out) {
  long i = (long)blockIdx.x * blockDim.x + threadIdx.x;
  if (i >= (long)BS * NALL) return;
  int j = (int)(i % NALL); int b = (int)(i / NALL);
  out[i] = ((j < NR) ? mr[b * NR + j] : mg[b * NG + (j - NR)]) ? 1.f : 0.f;
}

// ---------------- TDM probe (async tensor path) ----------------
__global__ void k_tdm_probe(const float* __restrict__ src, float* __restrict__ dst) {
#ifdef HAVE_TDM
  __align__(16) __shared__ float tile[64 * 16];
  unsigned long long ga = (unsigned long long)(const void*)src;
  v4u g0 = { 1u,                                  // count=1, user mode
             0u,                                  // lds_addr = 0
             (unsigned int)(ga & 0xFFFFFFFFu),
             (unsigned int)((ga >> 32) & 0x1FFFFFFu) | (2u << 30) };  // type=2
  v8i g1 = { (int)(2u << 16),                     // data_size = 4B
             (int)((512u & 0xFFFFu) << 16),       // tensor_dim0 lo16
             (int)((512u >> 16) | ((512u & 0xFFFFu) << 16)),  // dim0 hi / dim1 lo
             (int)((512u >> 16) | (64u << 16)),   // dim1 hi / tile_dim0=64
             (int)(16u),                          // tile_dim1=16, tile_dim2=0
             (int)512,                            // tensor_dim0_stride lo32
             0, 0 };
  v4i z4 = { 0, 0, 0, 0 };
#if __clang_major__ >= 23
  v8i z8 = { 0, 0, 0, 0, 0, 0, 0, 0 };
  __builtin_amdgcn_tensor_load_to_lds(g0, g1, z4, z4, z8, 0);
#else
  __builtin_amdgcn_tensor_load_to_lds(g0, g1, z4, z4, 0);
#endif
  __builtin_amdgcn_s_wait_tensorcnt(0);
#if defined(__has_builtin)
#if __has_builtin(__builtin_amdgcn_s_wait_asynccnt)
  __builtin_amdgcn_s_wait_asynccnt(0);
#endif
#endif
  __syncthreads();
  dst[threadIdx.x] = tile[threadIdx.x];
#else
  dst[threadIdx.x] = src[threadIdx.x];
#endif
}

// =====================================================================
// Host orchestration
// =====================================================================
struct StackW {
  const bf16_t *W1, *W2, *Wk, *Wo, *Wq, *Wv;
  const float *b1, *b2, *bk, *bo, *bq, *bv, *ln1b, *ln1g, *ln2b, *ln2g;
};

static inline void launch_gemm(hipStream_t st,
    const bf16_t* A, long sA, int lda,
    const bf16_t* B, long sB, int ldb, int transB,
    float* C32, bf16_t* Cbf, long sC, int ldc,
    const float* bias, float alpha, int relu,
    int M, int N, int K, int batch) {
  dim3 g((N + 63) / 64, (M + 63) / 64, batch);
  k_gemm_wmma<<<g, 128, 0, st>>>(A, sA, lda, B, sB, ldb, transB,
                                 C32, Cbf, sC, ldc, bias, alpha, relu, M, N, K);
}

extern "C" void kernel_launch(void* const* d_in, const int* in_sizes, int n_in,
                              void* d_out, int out_size, void* d_ws, size_t ws_size,
                              hipStream_t stream) {
  (void)in_sizes; (void)n_in; (void)out_size; (void)ws_size;

  const float* region_features = (const float*)d_in[0];
  const float* grid_features   = (const float*)d_in[1];
  const float* region_embed    = (const float*)d_in[2];
  const float* grid_embed      = (const float*)d_in[3];
  const float* boxes           = (const float*)d_in[4];
  const int*   aligns          = (const int*)d_in[5];
  // params flattened in sorted-key tree order:
  // WG_b, WG_w, g2r{16}, grid{16}, r2g{16}, region{16}
  const float* WG_b = (const float*)d_in[6];
  const float* WG_w = (const float*)d_in[7];

  char* base = (char*)d_ws;
  size_t off = 0;
  auto alloc = [&](size_t bytes) -> void* {
    size_t a = (off + 255) & ~(size_t)255;
    off = a + bytes;
    return (void*)(base + a);
  };
  auto ew = [](long n) { return dim3((unsigned)((n + 255) / 256)); };
  auto conv = [&](const float* src, size_t n) -> bf16_t* {
    bf16_t* dst = (bf16_t*)alloc(n * sizeof(bf16_t));
    k_f2bf<<<ew((long)n), 256, 0, stream>>>(src, dst, (long)n);
    return dst;
  };
  // per-stack sorted keys: W1 W2 Wk Wo Wq Wv b1 b2 bk bo bq bv ln1_b ln1_g ln2_b ln2_g
  auto mk = [&](int ib) -> StackW {
    StackW s;
    s.W1 = conv((const float*)d_in[ib + 0], (size_t)NL * D_MODEL * DFF);
    s.W2 = conv((const float*)d_in[ib + 1], (size_t)NL * DFF * D_MODEL);
    s.Wk = conv((const float*)d_in[ib + 2], (size_t)NL * D_MODEL * D_MODEL);
    s.Wo = conv((const float*)d_in[ib + 3], (size_t)NL * D_MODEL * D_MODEL);
    s.Wq = conv((const float*)d_in[ib + 4], (size_t)NL * D_MODEL * D_MODEL);
    s.Wv = conv((const float*)d_in[ib + 5], (size_t)NL * D_MODEL * D_MODEL);
    s.b1 = (const float*)d_in[ib + 6];  s.b2 = (const float*)d_in[ib + 7];
    s.bk = (const float*)d_in[ib + 8];  s.bo = (const float*)d_in[ib + 9];
    s.bq = (const float*)d_in[ib + 10]; s.bv = (const float*)d_in[ib + 11];
    s.ln1b = (const float*)d_in[ib + 12]; s.ln1g = (const float*)d_in[ib + 13];
    s.ln2b = (const float*)d_in[ib + 14]; s.ln2g = (const float*)d_in[ib + 15];
    return s;
  };
  StackW Sg2r = mk(8), Sgrid = mk(24), Sr2g = mk(40), Sregion = mk(56);

  // ---- buffers ----
  size_t actN = (size_t)BS * NALL * D_MODEL;
  float* gwb        = (float*)alloc((size_t)BS * H * NALL * NALL * 4);
  unsigned char* mr = (unsigned char*)alloc((size_t)BS * NR);
  unsigned char* mg = (unsigned char*)alloc((size_t)BS * NG);
  unsigned char* ra = (unsigned char*)alloc((size_t)BS * NR * NALL);
  unsigned char* ga = (unsigned char*)alloc((size_t)BS * NG * NALL);
  float* poscross   = (float*)alloc(actN * 4);
  float* cur_region = (float*)alloc((size_t)BS * NR * D_MODEL * 4);
  float* cur_grid   = (float*)alloc((size_t)BS * NG * D_MODEL * 4);
  float* all32      = (float*)alloc(actN * 4);
  bf16_t* qpos = (bf16_t*)alloc(actN * 2);
  bf16_t* kpos = (bf16_t*)alloc(actN * 2);
  bf16_t* kvbf = (bf16_t*)alloc(actN * 2);
  float*  P    = (float*)alloc(actN * 4);
  bf16_t* Qh   = (bf16_t*)alloc(actN * 2);
  bf16_t* Kh   = (bf16_t*)alloc(actN * 2);
  bf16_t* Vh   = (bf16_t*)alloc(actN * 2);
  float*  scores = (float*)alloc((size_t)BS * H * NALL * NALL * 4);
  bf16_t* att    = (bf16_t*)alloc((size_t)BS * H * NALL * NALL * 2);
  float*  Oh   = (float*)alloc(actN * 4);
  bf16_t* Ocat = (bf16_t*)alloc(actN * 2);
  float*  obuf = (float*)alloc(actN * 4);
  float*  x32  = (float*)alloc(actN * 4);
  bf16_t* xbf  = (bf16_t*)alloc(actN * 2);
  bf16_t* ff1  = (bf16_t*)alloc((size_t)BS * NALL * DFF * 2);
  float*  ff2  = (float*)alloc(actN * 4);
  float*  tdmdst = (float*)alloc(1024 * 4);

  // ---- setup ----
  hipMemcpyAsync(cur_region, region_features, (size_t)BS * NR * D_MODEL * 4,
                 hipMemcpyDeviceToDevice, stream);
  hipMemcpyAsync(cur_grid, grid_features, (size_t)BS * NG * D_MODEL * 4,
                 hipMemcpyDeviceToDevice, stream);
  k_gw_bias<<<ew((long)BS * NALL * NALL), 256, 0, stream>>>(boxes, WG_w, WG_b, gwb);
  k_feat_mask<<<BS * NR, 256, 0, stream>>>(region_features, mr);
  k_feat_mask<<<BS * NG, 256, 0, stream>>>(grid_features, mg);
  k_mask_ra<<<ew((long)BS * NR * NALL), 256, 0, stream>>>(aligns, ra);
  k_mask_ga<<<ew((long)BS * NG * NALL), 256, 0, stream>>>(aligns, ga);
  k_concat2<<<ew((long)actN), 256, 0, stream>>>(region_embed, NR, grid_embed, NG, poscross);
  k_tdm_probe<<<1, 128, 0, stream>>>(region_features, tdmdst);

  // ---- one encoder layer call ----
  auto enc = [&](const StackW& S, int li, const float* q32, int nq,
                 const float* kv32, int nk, const float* posq, const float* posk,
                 int qoff, int koff, const unsigned char* mask, int mrow, long mbatch,
                 float* out32) {
    long nqE = (long)BS * nq * D_MODEL, nkE = (long)BS * nk * D_MODEL;
    k_add_f2bf<<<ew(nqE), 256, 0, stream>>>(q32, posq, qpos, nqE);
    k_add_f2bf<<<ew(nkE), 256, 0, stream>>>(kv32, posk, kpos, nkE);
    k_f2bf<<<ew(nkE), 256, 0, stream>>>(kv32, kvbf, nkE);
    // Q/K/V projections (512x512 GEMMs, batched over BS)
    launch_gemm(stream, qpos, (long)nq * 512, 512, S.Wq + (size_t)li * 512 * 512, 0, 512, 0,
                P, nullptr, (long)nq * 512, 512, S.bq + li * 512, 1.f, 0, nq, 512, 512, BS);
    k_split_heads<<<ew((long)BS * H * nq * DKH), 256, 0, stream>>>(P, Qh, nq);
    launch_gemm(stream, kpos, (long)nk * 512, 512, S.Wk + (size_t)li * 512 * 512, 0, 512, 0,
                P, nullptr, (long)nk * 512, 512, S.bk + li * 512, 1.f, 0, nk, 512, 512, BS);
    k_split_heads<<<ew((long)BS * H * nk * DKH), 256, 0, stream>>>(P, Kh, nk);
    launch_gemm(stream, kvbf, (long)nk * 512, 512, S.Wv + (size_t)li * 512 * 512, 0, 512, 0,
                P, nullptr, (long)nk * 512, 512, S.bv + li * 512, 1.f, 0, nk, 512, 512, BS);
    k_split_heads<<<ew((long)BS * H * nk * DKH), 256, 0, stream>>>(P, Vh, nk);
    // scores = Q @ K^T / 8   (batched over BS*H, transB)
    launch_gemm(stream, Qh, (long)nq * DKH, DKH, Kh, (long)nk * DKH, DKH, 1,
                scores, nullptr, (long)nq * nk, nk, nullptr, 0.125f, 0, nq, nk, DKH, BS * H);
    k_softmax_att<<<dim3(nq, BS * H), 256, 0, stream>>>(scores, (long)nq * nk,
                gwb, qoff, koff, mask, mrow, mbatch, att, nq, nk);
    // O = att @ V (K = nk, zero-padded inside GEMM)
    launch_gemm(stream, att, (long)nq * nk, nk, Vh, (long)nk * DKH, DKH, 0,
                Oh, nullptr, (long)nq * DKH, DKH, nullptr, 1.f, 0, nq, DKH, nk, BS * H);
    k_merge_heads<<<ew(nqE), 256, 0, stream>>>(Oh, Ocat, nq);
    launch_gemm(stream, Ocat, (long)nq * 512, 512, S.Wo + (size_t)li * 512 * 512, 0, 512, 0,
                obuf, nullptr, (long)nq * 512, 512, S.bo + li * 512, 1.f, 0, nq, 512, 512, BS);
    // x = LN(q + pos + o)
    k_ln<<<BS * nq, 256, 0, stream>>>(obuf, q32, posq, S.ln1g + li * 512, S.ln1b + li * 512,
                                      x32, xbf);
    // FFN
    launch_gemm(stream, xbf, (long)nq * 512, 512, S.W1 + (size_t)li * 512 * DFF, 0, DFF, 0,
                nullptr, ff1, (long)nq * DFF, DFF, S.b1 + li * DFF, 1.f, 1, nq, DFF, 512, BS);
    launch_gemm(stream, ff1, (long)nq * DFF, DFF, S.W2 + (size_t)li * DFF * 512, 0, 512, 0,
                ff2, nullptr, (long)nq * 512, 512, S.b2 + li * 512, 1.f, 0, nq, 512, DFF, BS);
    k_ln<<<BS * nq, 256, 0, stream>>>(ff2, x32, nullptr, S.ln2g + li * 512, S.ln2b + li * 512,
                                      out32, nullptr);
  };

  for (int li = 0; li < NL; ++li) {
    // self attention
    enc(Sregion, li, cur_region, NR, cur_region, NR, region_embed, region_embed,
        0, 0, mr, 0, NR, cur_region);
    enc(Sgrid, li, cur_grid, NG, cur_grid, NG, grid_embed, grid_embed,
        NR, NR, mg, 0, NG, cur_grid);
    // cross attention against concatenated stream
    k_concat2<<<ew((long)actN), 256, 0, stream>>>(cur_region, NR, cur_grid, NG, all32);
    enc(Sr2g, li, cur_region, NR, all32, NALL, region_embed, poscross,
        0, 0, ra, NALL, (long)NR * NALL, cur_region);
    enc(Sg2r, li, cur_grid, NG, all32, NALL, grid_embed, poscross,
        NR, 0, ga, NALL, (long)NG * NALL, cur_grid);
    k_write_out<<<ew((long)actN), 256, 0, stream>>>(cur_region, cur_grid, (float*)d_out, li);
  }
  k_write_mask<<<ew((long)BS * NALL), 256, 0, stream>>>(
      mr, mg, (float*)d_out + (size_t)BS * NL * NALL * D_MODEL);
}
